// CosineVectorQuantizer_30039001268974
// MI455X (gfx1250) — matrix operand order repaired and checked by reading, not compiled, not run on hardware
//
#include <hip/hip_runtime.h>
#include <hip/hip_bf16.h>
#include <math.h>

typedef __attribute__((ext_vector_type(16))) _Float16 v16h;
typedef __attribute__((ext_vector_type(8)))  float    v8f;

#define DDIM 128
#define TILE_N 16
#define WAVES_PER_BLOCK 8
#define ROWS_PER_WAVE 16
#define ROWS_PER_BLOCK (WAVES_PER_BLOCK * ROWS_PER_WAVE)

// ---------------------------------------------------------------------------
// Kernel 0: L2-normalize the codebook rows, emit f16 copy for the WMMA GEMM.
// ---------------------------------------------------------------------------
__global__ __launch_bounds__(DDIM) void vq_normalize_codebook(
    const float* __restrict__ w, _Float16* __restrict__ cbn) {
  const int row = blockIdx.x;
  const int t = threadIdx.x;              // 0..127
  const float v = w[(size_t)row * DDIM + t];
  float s = v * v;
#pragma unroll
  for (int m = 16; m >= 1; m >>= 1) s += __shfl_xor(s, m, 32);
  __shared__ float ssum[4];
  if ((t & 31) == 0) ssum[t >> 5] = s;
  __syncthreads();
  const float tot = ssum[0] + ssum[1] + ssum[2] + ssum[3];
  const float inv = 1.0f / fmaxf(sqrtf(tot), 1e-12f);
  cbn[(size_t)row * DDIM + t] = (_Float16)(v * inv);
}

// ---------------------------------------------------------------------------
// Kernel 1: fused sim-GEMM (f16 WMMA) + argmax + projection + loss partials.
// Each wave owns 16 rows of x; block owns 128 rows. A = raw x (positive row
// scaling doesn't change argmax). Inner loop is software-pipelined: tile
// nt+1's B loads are in flight while tile nt's 4 WMMAs execute.
// ---------------------------------------------------------------------------
__global__ __launch_bounds__(256) void vq_main(
    const float* __restrict__ x, const float* __restrict__ emb,
    const _Float16* __restrict__ cbn,
    float* __restrict__ out_xq, float* __restrict__ out_idx,
    float* __restrict__ out_scal, float* __restrict__ partial,
    int NE) {
  const int lane = threadIdx.x & 31;
  const int wave = threadIdx.x >> 5;
  const int lo   = lane & 15;             // row-within-tile (A) / column (B)
  const int hi   = lane >> 4;             // which K half-group this lane holds

  // ---- A operands: 16 rows x K=128 of x in f16, contiguous K layout
  // (lanes 0-15 hold K c*32+0..15; lanes 16-31 hold K c*32+16..31).
  const int rowA = blockIdx.x * ROWS_PER_BLOCK + wave * ROWS_PER_WAVE + lo;
  const float* __restrict__ xr = x + (size_t)rowA * DDIM;
  v16h Aop[4];
#pragma unroll
  for (int c = 0; c < 4; ++c) {
    const float* p0 = xr + c * 32 + hi * 16;
    const float* p1 = p0 + 8;
#pragma unroll
    for (int j = 0; j < 8; ++j) {
      Aop[c][j]     = (_Float16)p0[j];
      Aop[c][8 + j] = (_Float16)p1[j];
    }
  }

  float bestv[8];
  int   besti[8];
#pragma unroll
  for (int s = 0; s < 8; ++s) { bestv[s] = -INFINITY; besti[s] = 0; }

  // Lane-invariant parts of the B address: lane's column row = nt*16+lo,
  // contiguous 16 halves per K-chunk (B = cbn^T maps to row-major cbn rows).
  const _Float16* __restrict__ brL = cbn + (size_t)lo * DDIM + hi * 16;
  const size_t tileStride = (size_t)TILE_N * DDIM;   // halves per 16-col tile

  auto loadB = [&](v16h* Bv, int nt) {
    const _Float16* br = brL + (size_t)nt * tileStride;
#pragma unroll
    for (int c = 0; c < 4; ++c) Bv[c] = *(const v16h*)(br + c * 32);
  };
  auto scoreTile = [&](const v16h* Bv, int nt) {
    v8f acc = {};
#pragma unroll
    for (int c = 0; c < 4; ++c) {
      acc = __builtin_amdgcn_wmma_f32_16x16x32_f16(
          false, Aop[c], false, Bv[c], (short)0, acc, false, false);
    }
    const int ncol = nt * TILE_N + lo;
#pragma unroll
    for (int s = 0; s < 8; ++s) {
      const float v = acc[s];
      if (v > bestv[s]) { bestv[s] = v; besti[s] = ncol; }  // first-max kept
    }
  };

  // ---- Two-stage pipelined sweep over all codebook tiles (ntiles even).
  const int ntiles = NE / TILE_N;
  v16h B0[4], B1[4];
  loadB(B0, 0);
  for (int nt = 0; nt < ntiles; nt += 2) {
    loadB(B1, nt + 1);                                  // in flight over B0's WMMAs
    const int pf = (nt + 3 < ntiles) ? nt + 3 : ntiles - 1;
    __builtin_prefetch(brL + (size_t)pf * tileStride, 0, 3);  // near prefetch
    scoreTile(B0, nt);
    const int nt2 = (nt + 2 < ntiles) ? nt + 2 : ntiles - 1;
    loadB(B0, nt2);                                     // in flight over B1's WMMAs
    scoreTile(B1, nt + 1);
  }

  // ---- Per-row epilogue: cross-lane argmax, projection, outputs, loss.
  float lossAcc = 0.0f;
#pragma unroll
  for (int s = 0; s < 8; ++s) {
    float bv = bestv[s];
    int   bi = besti[s];
#pragma unroll
    for (int m = 8; m >= 1; m >>= 1) {   // butterfly within 16-lane group
      const float ov = __shfl_xor(bv, m, 32);
      const int   oi = __shfl_xor(bi, m, 32);
      if (ov > bv || (ov == bv && oi < bi)) { bv = ov; bi = oi; }
    }
    const int row = blockIdx.x * ROWS_PER_BLOCK + wave * ROWS_PER_WAVE
                    + hi * 8 + s;
    const int d0 = lo * 8;               // 16 lanes x 8 floats = D=128
    const float* __restrict__ ep = emb + (size_t)bi * DDIM + d0;
    const float* __restrict__ xp = x + (size_t)row * DDIM + d0;
    float ev[8], xv[8];
#pragma unroll
    for (int j = 0; j < 8; ++j) { ev[j] = ep[j]; xv[j] = xp[j]; }
    float pd = 0.0f, pn = 0.0f;
#pragma unroll
    for (int j = 0; j < 8; ++j) { pd += xv[j] * ev[j]; pn += ev[j] * ev[j]; }
#pragma unroll
    for (int m = 8; m >= 1; m >>= 1) {
      pd += __shfl_xor(pd, m, 32);
      pn += __shfl_xor(pn, m, 32);
    }
    const float scal = pd / (pn + 1e-8f);
    float* __restrict__ op = out_xq + (size_t)row * DDIM + d0;
#pragma unroll
    for (int j = 0; j < 8; ++j) {
      const float p   = scal * ev[j];
      const float dlt = p - xv[j];
      op[j] = xv[j] + dlt;               // x + (proj - x), as reference writes
      lossAcc += dlt * dlt;
    }
    if (lo == 0) {
      out_idx[row]  = (float)bi;
      out_scal[row] = scal;
    }
  }

  // Deterministic loss partial: wave butterfly -> fixed-order block sum.
#pragma unroll
  for (int m = 16; m >= 1; m >>= 1) lossAcc += __shfl_xor(lossAcc, m, 32);
  __shared__ float wl[WAVES_PER_BLOCK];
  if (lane == 0) wl[wave] = lossAcc;
  __syncthreads();
  if (threadIdx.x == 0) {
    float t = 0.0f;
#pragma unroll
    for (int w = 0; w < WAVES_PER_BLOCK; ++w) t += wl[w];
    partial[blockIdx.x] = t;
  }
}

// ---------------------------------------------------------------------------
// Kernel 2: deterministic fixed-tree reduction of per-block loss partials.
// ---------------------------------------------------------------------------
__global__ __launch_bounds__(256) void vq_reduce_loss(
    const float* __restrict__ partial, int n,
    float* __restrict__ out_loss, float scale) {
  __shared__ float sm[256];
  float a = 0.0f;
  for (int i = threadIdx.x; i < n; i += 256) a += partial[i];
  sm[threadIdx.x] = a;
  __syncthreads();
  for (int s = 128; s >= 1; s >>= 1) {
    if (threadIdx.x < s) sm[threadIdx.x] += sm[threadIdx.x + s];
    __syncthreads();
  }
  if (threadIdx.x == 0) *out_loss = sm[0] * scale;
}

// ---------------------------------------------------------------------------
extern "C" void kernel_launch(void* const* d_in, const int* in_sizes, int n_in,
                              void* d_out, int out_size, void* d_ws,
                              size_t ws_size, hipStream_t stream) {
  (void)n_in; (void)out_size; (void)ws_size;
  const float* x   = (const float*)d_in[0];
  const float* emb = (const float*)d_in[1];
  const int B  = in_sizes[0] / DDIM;     // 131072
  const int NE = in_sizes[1] / DDIM;     // 2048

  _Float16* cbn = (_Float16*)d_ws;                       // NE*D f16 = 512 KB
  size_t cbnBytes = (size_t)NE * DDIM * sizeof(_Float16);
  float* partial = (float*)((char*)d_ws + ((cbnBytes + 255) & ~(size_t)255));

  float* out_xq   = (float*)d_out;                       // [B, D]
  float* out_loss = out_xq + (size_t)B * DDIM;           // [1]
  float* out_idx  = out_loss + 1;                        // [B] (as float)
  float* out_scal = out_idx + B;                         // [B]

  const int nBlocks = B / ROWS_PER_BLOCK;                // 1024

  vq_normalize_codebook<<<NE, DDIM, 0, stream>>>(emb, cbn);
  vq_main<<<nBlocks, 256, 0, stream>>>(x, emb, cbn, out_xq, out_idx,
                                       out_scal, partial, NE);
  vq_reduce_loss<<<1, 256, 0, stream>>>(
      partial, nBlocks, out_loss,
      1.25f / (float)((size_t)B * DDIM));   // codebook + 0.25*commitment
}
